// ResMambaBlock_34823594836273
// MI455X (gfx1250) — matrix operand order repaired and checked by reading, not compiled, not run on hardware
//
#include <hip/hip_runtime.h>
#include <hip/hip_bf16.h>

// ---------------- problem constants ----------------
constexpr int BB  = 2;
constexpr int CC  = 64;
constexpr int HH  = 96;
constexpr int WW  = 96;
constexpr int LL  = HH * WW;      // 9216 tokens per batch (= GEMM N, constexpr)
constexpr int DI  = 128;
constexpr int DS  = 16;
constexpr int DTR = 4;
constexpr int DC  = 4;
constexpr float EPS = 1e-5f;

typedef __attribute__((ext_vector_type(16))) _Float16 v16h;
typedef __attribute__((ext_vector_type(8)))  float    v8f;

// =====================================================================
// WMMA GEMM:  Out[m,n] = act( sum_k W[m,k] * X[k,n] + bias[m] ) + sc*Add[m,n]
// W row-major (M x KT), X/Out/Add channels-major (KT x N)/(M x N), batched
// via z-dim.  A block covers MT m-tiles x 128 columns (8 waves x 16 cols).
// N and KT are compile-time: all B-fragment loads become base+immediate
// (no 64-bit add chains), and the K loop fully unrolls.
// Per K-step: 256 threads stage the (MT*16) x 32 weight slab into LDS in
// WMMA-fragment order (f16, ds_store_b32), then each wave pulls its per-lane
// A fragments back with 32-byte LDS loads and issues MT back-to-back
// v_wmma_f32_16x16x32_f16 against one B fragment.
// =====================================================================
template<int MT, int KT, bool MASK>
__global__ void wmma_gemm_kernel(const float* __restrict__ Wt, int M,
                                 const float* __restrict__ X,  long long xStride,
                                 float* __restrict__ Out,      long long oStride,
                                 const float* __restrict__ Add, long long aStride,
                                 const float* __restrict__ bias,
                                 const float* __restrict__ scale,  // device scalar or null
                                 int relu)
{
    constexpr int N = LL;
    // MT*32 fragments, 16 halves each, fragment-contiguous (32B aligned)
    __shared__ __align__(32) _Float16 aLds[MT * 32 * 16];
    unsigned* aLdsU = (unsigned*)aLds;

    const int lane = threadIdx.x & 31;
    const int wave = threadIdx.x >> 5;
    const int half = lane >> 4;      // 0: lanes 0-15, 1: lanes 16-31
    const int l16  = lane & 15;

    const int nb   = blockIdx.x * 128 + wave * 16;   // N tile base (this wave)
    const int mb   = blockIdx.y * (MT * 16);         // M slab base (this block)
    const int ncol = nb + l16;

    const float* Xb = X + (long long)blockIdx.z * xStride;
    float*       Ob = Out + (long long)blockIdx.z * oStride;
    const float* Ab = Add ? Add + (long long)blockIdx.z * aStride : nullptr;
    const float  sc = scale ? *scale : 1.0f;

    // ---- per-thread A staging descriptors (fixed across K steps)
    int       stSlot[MT];   // packed-pair slot in LDS
    long long stBase[MT];   // element offset of (row, k0) in W
#pragma unroll
    for (int it = 0; it < MT; ++it) {
        const int slot = it * 256 + threadIdx.x;
        const int frag = slot >> 3;                // fragment 0..MT*32-1
        const int i    = slot & 7;                 // pair index in fragment
        const int fl   = frag & 31;                // fragment's lane
        const int mt   = frag >> 5;                // fragment's m-tile
        const int m    = mb + mt * 16 + (fl & 15);
        const int mc   = MASK ? ((m < M) ? m : 0) : m;  // clamp; junk rows masked at store
        const int k0   = ((fl >> 4) << 4) + (i << 1);
        stSlot[it] = frag * 8 + i;
        stBase[it] = (long long)mc * KT + k0;
    }
    // B-fragment column base: Xcol[(kk + 2j)*N] are immediate offsets
    const float* Xcol = Xb + half * 16 * N + ncol;

    v8f acc[MT];
#pragma unroll
    for (int t = 0; t < MT; ++t) acc[t] = v8f{};

#pragma unroll
    for (int kk = 0; kk < KT; kk += 32) {
        // ---- cooperative A staging: MT*256 packed f16 pairs
#pragma unroll
        for (int it = 0; it < MT; ++it) {
            const float2 w2 = *(const float2*)(Wt + stBase[it] + kk);
            union { _Float16 h[2]; unsigned u; } pk;
            pk.h[0] = (_Float16)w2.x;
            pk.h[1] = (_Float16)w2.y;
            aLdsU[stSlot[it]] = pk.u;
        }
        __syncthreads();

        // ---- B fragment (32x16 f16): lanes 0-15 K=kk..kk+15, lanes 16-31
        //      K=kk+16..kk+31; VGPR j packs K=(2j,2j+1) for column N=lane&15
        v16h bf;
#pragma unroll
        for (int j = 0; j < 8; ++j) {
            bf[2 * j]     = (_Float16)Xcol[(kk + 2 * j)     * N];
            bf[2 * j + 1] = (_Float16)Xcol[(kk + 2 * j + 1) * N];
        }
        if (kk + 32 < KT)   // pull next K-slab of X toward the WGP (global_prefetch_b8)
            __builtin_prefetch(&Xcol[(kk + 32) * N], 0, 1);

        // ---- MT WMMAs against one B fragment; A straight from LDS
#pragma unroll
        for (int mt = 0; mt < MT; ++mt) {
            const v16h a = *(const v16h*)&aLds[(mt * 32 + lane) * 16];
            acc[mt] = __builtin_amdgcn_wmma_f32_16x16x32_f16(
                          false, a, false, bf, (short)0, acc[mt], false, false);
        }
        __syncthreads();
    }

    // ---- D store: tile row = base + r + half*8, col = nb + (lane&15)
#pragma unroll
    for (int mt = 0; mt < MT; ++mt) {
#pragma unroll
        for (int r = 0; r < 8; ++r) {
            const int row = mb + mt * 16 + r + half * 8;
            if (!MASK || row < M) {
                float v = acc[mt][r];
                if (bias) v += bias[row];
                if (relu) v = fmaxf(v, 0.0f);
                if (Ab)   v += sc * Ab[row * N + ncol];
                Ob[row * N + ncol] = v;
            }
        }
    }
}

// =====================================================================
// Instance norm over one (b,c) plane of HW elements, optional ReLU.
// =====================================================================
__global__ void instnorm_kernel(const float* __restrict__ in,
                                float* __restrict__ out, int HWn, int relu)
{
    __shared__ float s1[256], s2[256];
    const long long base = (long long)blockIdx.x * HWn;
    float sum = 0.f, sq = 0.f;
    for (int i = threadIdx.x; i < HWn; i += blockDim.x) {
        float v = in[base + i];
        sum += v; sq += v * v;
    }
    s1[threadIdx.x] = sum; s2[threadIdx.x] = sq;
    __syncthreads();
    for (int o = 128; o > 0; o >>= 1) {
        if (threadIdx.x < o) {
            s1[threadIdx.x] += s1[threadIdx.x + o];
            s2[threadIdx.x] += s2[threadIdx.x + o];
        }
        __syncthreads();
    }
    const float mu   = s1[0] / HWn;
    const float var  = s2[0] / HWn - mu * mu;
    const float rstd = rsqrtf(var + EPS);
    for (int i = threadIdx.x; i < HWn; i += blockDim.x) {
        float v = (in[base + i] - mu) * rstd;
        if (relu) v = fmaxf(v, 0.0f);
        out[base + i] = v;
    }
}

// =====================================================================
// 3x3 depthwise conv, zero pad 1, + per-channel bias.  (b,c,h,w) layout.
// =====================================================================
__global__ void dwconv3x3_kernel(const float* __restrict__ in,
                                 const float* __restrict__ w,
                                 const float* __restrict__ bias,
                                 float* __restrict__ out)
{
    const int idx = blockIdx.x * blockDim.x + threadIdx.x;
    if (idx >= BB * CC * LL) return;
    const int px = idx % WW;
    int t = idx / WW;
    const int py = t % HH; t /= HH;
    const int c  = t % CC;
    const float* pin = in + (long long)(idx / LL) * LL;
    const float* pw  = w + c * 9;
    float s = bias[c];
#pragma unroll
    for (int ky = 0; ky < 3; ++ky) {
        const int yy = py + ky - 1;
        if (yy < 0 || yy >= HH) continue;
#pragma unroll
        for (int kx = 0; kx < 3; ++kx) {
            const int xx = px + kx - 1;
            if (xx < 0 || xx >= WW) continue;
            s += pw[ky * 3 + kx] * pin[yy * WW + xx];
        }
    }
    out[idx] = s;
}

// =====================================================================
// LayerNorm over C channels per token; data in (b,c,l) layout.
// =====================================================================
__global__ void layernorm_kernel(const float* __restrict__ in,
                                 const float* __restrict__ g,
                                 const float* __restrict__ bt,
                                 float* __restrict__ out)
{
    const int idx = blockIdx.x * blockDim.x + threadIdx.x;
    if (idx >= BB * LL) return;
    const int b = idx / LL, l = idx % LL;
    const float* p = in + (long long)b * CC * LL + l;
    float sum = 0.f, sq = 0.f;
#pragma unroll 4
    for (int c = 0; c < CC; ++c) {
        float v = p[c * LL];
        sum += v; sq += v * v;
    }
    const float mu   = sum / CC;
    const float rstd = rsqrtf(sq / CC - mu * mu + EPS);
    float* o = out + (long long)b * CC * LL + l;
#pragma unroll 4
    for (int c = 0; c < CC; ++c)
        o[c * LL] = (p[c * LL] - mu) * rstd * g[c] + bt[c];
}

// =====================================================================
// Causal depthwise conv1d (width 4, left pad 3) + bias + SiLU.
// xz is (b, 2*DI, L); we read rows 0..DI-1.  xc out is (b, DI, L).
// =====================================================================
__global__ void conv1d_silu_kernel(const float* __restrict__ xz,
                                   const float* __restrict__ cw,
                                   const float* __restrict__ cb,
                                   float* __restrict__ xc)
{
    const int idx = blockIdx.x * blockDim.x + threadIdx.x;
    if (idx >= BB * DI * LL) return;
    const int l = idx % LL;
    int t = idx / LL;
    const int d = t % DI, b = t / DI;
    const float* xin = xz + ((long long)b * 2 * DI + d) * LL;
    float s = cb[d];
#pragma unroll
    for (int j = 0; j < DC; ++j) {
        const int ll = l - (DC - 1) + j;
        if (ll >= 0) s += cw[d * DC + j] * xin[ll];
    }
    const float sig = 1.0f / (1.0f + expf(-s));
    xc[((long long)b * DI + d) * LL + l] = s * sig;
}

// =====================================================================
// delta = softplus(dt_proj_w @ dbl[0:4] + dt_proj_b);  dbl is (b,36,L).
// =====================================================================
__global__ void dt_softplus_kernel(const float* __restrict__ dbl,
                                   const float* __restrict__ dtw,
                                   const float* __restrict__ dtb,
                                   float* __restrict__ delta)
{
    const int idx = blockIdx.x * blockDim.x + threadIdx.x;
    if (idx >= BB * DI * LL) return;
    const int l = idx % LL;
    int t = idx / LL;
    const int d = t % DI, b = t / DI;
    const float* db = dbl + (long long)b * (DTR + 2 * DS) * LL;
    float s = dtb[d];
#pragma unroll
    for (int r = 0; r < DTR; ++r) s += dtw[d * DTR + r] * db[r * LL + l];
    delta[idx] = (s > 20.0f) ? s : log1pf(expf(s));
}

// =====================================================================
// Selective scan: one (b,d) pair per 16-lane half-wave, one state s per
// lane.  h = exp(delta*A)*h + delta*B*xc ; y = sum_s h*C (shfl_xor tree),
// fused with  y = (y + D*xc) * silu(z).  Grid: 16 blocks x 256 threads.
// =====================================================================
__global__ void scan_kernel(const float* __restrict__ delta,
                            const float* __restrict__ dbl,
                            const float* __restrict__ xc,
                            const float* __restrict__ xz,
                            const float* __restrict__ Alog,
                            const float* __restrict__ Dp,
                            float* __restrict__ y)
{
    const int s  = threadIdx.x & 15;
    const int p  = blockIdx.x * 16 + (threadIdx.x >> 4);   // 0..255
    const int b  = p / DI, d = p % DI;

    const float* dblb = dbl + (long long)b * (DTR + 2 * DS) * LL;
    const float* Brow = dblb + (DTR + s) * LL;
    const float* Crow = dblb + (DTR + DS + s) * LL;
    const float* drow = delta + ((long long)b * DI + d) * LL;
    const float* xrow = xc    + ((long long)b * DI + d) * LL;
    const float* zrow = xz    + ((long long)b * 2 * DI + DI + d) * LL;
    float*       yrow = y     + ((long long)b * DI + d) * LL;

    const float A  = -expf(Alog[d * DS + s]);
    const float Dd = Dp[d];
    float h = 0.0f;
    for (int l = 0; l < LL; ++l) {
        const float dl = drow[l];
        const float xv = xrow[l];
        h = expf(dl * A) * h + dl * Brow[l] * xv;
        float c = h * Crow[l];
        c += __shfl_xor(c, 1, 32);
        c += __shfl_xor(c, 2, 32);
        c += __shfl_xor(c, 4, 32);
        c += __shfl_xor(c, 8, 32);
        if (s == 0) {
            const float zv = zrow[l];
            yrow[l] = (c + Dd * xv) * (zv / (1.0f + expf(-zv)));
        }
    }
}

// =====================================================================
__global__ void add_kernel(const float* __restrict__ a,
                           const float* __restrict__ b,
                           float* __restrict__ out, int n)
{
    const int i = blockIdx.x * blockDim.x + threadIdx.x;
    if (i < n) out[i] = a[i] + b[i];
}

// ---------------- host-side helpers ----------------
static inline void gemm(const float* Wt, int M, int K,
                        const float* X, long long xs,
                        float* Out, long long os,
                        const float* Add, long long as,
                        const float* bias, const float* scale,
                        int relu, hipStream_t st)
{
    if (M == DTR + 2 * DS) {           // x_proj: M=36, K=128, masked
        dim3 g(LL / 128, 1, BB);
        wmma_gemm_kernel<3, 128, true><<<g, 256, 0, st>>>(
            Wt, M, X, xs, Out, os, Add, as, bias, scale, relu);
    } else if (K == 64) {              // pw*/proj/in_proj: M in {64,256}, K=64
        dim3 g(LL / 128, M / 64, BB);
        wmma_gemm_kernel<4, 64, false><<<g, 256, 0, st>>>(
            Wt, M, X, xs, Out, os, Add, as, bias, scale, relu);
    } else {                           // out_proj: M=64, K=128
        dim3 g(LL / 128, M / 64, BB);
        wmma_gemm_kernel<4, 128, false><<<g, 256, 0, st>>>(
            Wt, M, X, xs, Out, os, Add, as, bias, scale, relu);
    }
}

struct MambaP {
    const float *ln_w, *ln_b, *in_proj_w, *conv_w, *conv_b, *x_proj_w,
                *dt_proj_w, *dt_proj_b, *A_log, *D, *out_proj_w,
                *proj_w, *proj_b, *skip;
};

static void mamba_layer(const float* xin /*(b,64,L)*/, float* out,
                        float* ln, float* xz, float* xc, float* dbl,
                        float* delta, float* yb, float* xm,
                        const MambaP& P, hipStream_t st)
{
    const int nTok = BB * LL;
    const int nE   = BB * DI * LL;
    layernorm_kernel<<<(nTok + 255) / 256, 256, 0, st>>>(xin, P.ln_w, P.ln_b, ln);
    // in_proj: (256x64) @ (64,L)
    gemm(P.in_proj_w, 2 * DI, CC, ln, (long long)CC * LL,
         xz, (long long)2 * DI * LL, nullptr, 0, nullptr, nullptr, 0, st);
    conv1d_silu_kernel<<<(nE + 255) / 256, 256, 0, st>>>(xz, P.conv_w, P.conv_b, xc);
    // x_proj: (36x128) @ (128,L), row-masked in the GEMM store
    gemm(P.x_proj_w, DTR + 2 * DS, DI, xc, (long long)DI * LL,
         dbl, (long long)(DTR + 2 * DS) * LL, nullptr, 0, nullptr, nullptr, 0, st);
    dt_softplus_kernel<<<(nE + 255) / 256, 256, 0, st>>>(dbl, P.dt_proj_w, P.dt_proj_b, delta);
    scan_kernel<<<16, 256, 0, st>>>(delta, dbl, xc, xz, P.A_log, P.D, yb);
    // out_proj: (64x128) @ (128,L), fused  + skip * xf
    gemm(P.out_proj_w, CC, DI, yb, (long long)DI * LL,
         xm, (long long)CC * LL, xin, (long long)CC * LL, nullptr, P.skip, 0, st);
    layernorm_kernel<<<(nTok + 255) / 256, 256, 0, st>>>(xm, P.ln_w, P.ln_b, ln);
    // proj: (64x64) @ (64,L) + bias
    gemm(P.proj_w, CC, CC, ln, (long long)CC * LL,
         out, (long long)CC * LL, nullptr, 0, P.proj_b, nullptr, 0, st);
}

extern "C" void kernel_launch(void* const* d_in, const int* in_sizes, int n_in,
                              void* d_out, int out_size, void* d_ws, size_t ws_size,
                              hipStream_t stream)
{
    (void)in_sizes; (void)n_in; (void)out_size; (void)ws_size;
    // setup_inputs() dict (insertion) order: x, gsc{10}, m1{14}, m2{14}
    const float* x      = (const float*)d_in[0];
    const float* dw1_w  = (const float*)d_in[1];
    const float* dw1_b  = (const float*)d_in[2];
    const float* pw1_w  = (const float*)d_in[3];
    const float* pw1_b  = (const float*)d_in[4];
    const float* p2_w   = (const float*)d_in[5];
    const float* p2_b   = (const float*)d_in[6];
    const float* dw3_w  = (const float*)d_in[7];
    const float* dw3_b  = (const float*)d_in[8];
    const float* pw3_w  = (const float*)d_in[9];
    const float* pw3_b  = (const float*)d_in[10];
    auto mk = [&](int base) {
        MambaP P;
        P.ln_w = (const float*)d_in[base + 0];  P.ln_b = (const float*)d_in[base + 1];
        P.in_proj_w = (const float*)d_in[base + 2];
        P.conv_w = (const float*)d_in[base + 3]; P.conv_b = (const float*)d_in[base + 4];
        P.x_proj_w = (const float*)d_in[base + 5];
        P.dt_proj_w = (const float*)d_in[base + 6]; P.dt_proj_b = (const float*)d_in[base + 7];
        P.A_log = (const float*)d_in[base + 8]; P.D = (const float*)d_in[base + 9];
        P.out_proj_w = (const float*)d_in[base + 10];
        P.proj_w = (const float*)d_in[base + 11]; P.proj_b = (const float*)d_in[base + 12];
        P.skip = (const float*)d_in[base + 13];
        return P;
    };
    const MambaP m1 = mk(11), m2 = mk(25);

    // workspace layout, P = one (B,64,L) fp32 plane = 4.5 MiB
    const size_t P = (size_t)BB * CC * LL;
    float* ws   = (float*)d_ws;
    float* t0   = ws;            // instnorm out
    float* t1   = ws + 1 * P;    // dwconv out
    float* x1b  = ws + 2 * P;    // x1
    float* xsb  = ws + 3 * P;    // xs
    float* gout = ws + 4 * P;    // gsc out == identity
    float* fr   = ws + 5 * P;    // relu(instnorm) -> mamba input
    float* lnb  = ws + 6 * P;    // layernorm scratch
    float* xzb  = ws + 7 * P;    // (b,256,L)  -> 4P
    float* xcb  = ws + 11 * P;   // (b,128,L)  -> 2P
    float* dblb = ws + 13 * P;   // (b,36,L)   -> <1P
    float* dltb = ws + 14 * P;   // (b,128,L)  -> 2P
    float* ybuf = ws + 16 * P;   // (b,128,L)  -> 2P
    float* xmb  = ws + 18 * P;   // (b,64,L)
    float* mout = ws + 19 * P;   // mamba layer output

    const int nE = BB * CC * LL;

    // ---------------- GSC ----------------
    instnorm_kernel<<<BB * CC, 256, 0, stream>>>(x, t0, LL, 0);
    dwconv3x3_kernel<<<(nE + 255) / 256, 256, 0, stream>>>(t0, dw1_w, dw1_b, t1);
    gemm(pw1_w, CC, CC, t1, (long long)CC * LL, x1b, (long long)CC * LL,
         nullptr, 0, pw1_b, nullptr, /*relu=*/1, stream);                 // x1
    gemm(p2_w, CC, CC, t0, (long long)CC * LL, xsb, (long long)CC * LL,
         x1b, (long long)CC * LL, p2_b, nullptr, /*relu=*/1, stream);     // xs = relu(..)+x1
    instnorm_kernel<<<BB * CC, 256, 0, stream>>>(xsb, t0, LL, 0);
    dwconv3x3_kernel<<<(nE + 255) / 256, 256, 0, stream>>>(t0, dw3_w, dw3_b, t1);
    gemm(pw3_w, CC, CC, t1, (long long)CC * LL, gout, (long long)CC * LL,
         x, (long long)CC * LL, pw3_b, nullptr, /*relu=*/1, stream);      // x3 + res

    // ---------------- Mamba layer 1 ----------------
    instnorm_kernel<<<BB * CC, 256, 0, stream>>>(gout, fr, LL, 1);        // relu(IN(x))
    mamba_layer(fr, mout, lnb, xzb, xcb, dblb, dltb, ybuf, xmb, m1, stream);

    // ---------------- Mamba layer 2 ----------------
    instnorm_kernel<<<BB * CC, 256, 0, stream>>>(mout, fr, LL, 1);
    mamba_layer(fr, mout, lnb, xzb, xcb, dblb, dltb, ybuf, xmb, m2, stream);

    // ---------------- final residual ----------------
    add_kernel<<<(nE + 255) / 256, 256, 0, stream>>>(mout, gout, (float*)d_out, nE);
}
